// InfoNceLoss_16844861735543
// MI455X (gfx1250) — matrix operand order repaired and checked by reading, not compiled, not run on hardware
//
#include <hip/hip_runtime.h>
#include <hip/hip_bf16.h>
#include <math.h>

typedef __attribute__((ext_vector_type(16))) _Float16 v16h;
typedef __attribute__((ext_vector_type(8)))  _Float16 v8h;
typedef __attribute__((ext_vector_type(8)))  float    v8f;
typedef __attribute__((ext_vector_type(4)))  float    v4f;

#define N_ROWS    8192
#define DIM       1024
#define TEMP_INV  10.0f
#define EPSF      1e-8f
#define COL_SPLIT 2
#define M_STRIP   64
#define N_STEP    128
#define NSLOTS    8   // partial writers per row: 4 npair waves x 2 col splits

// ---------------------------------------------------------------------------
// Kernel 1: per-row L2 normalize (matches F.normalize: x / max(||x||, 1e-12))
// and convert to f16 for the WMMA GEMM. One 256-thread block per row.
// ---------------------------------------------------------------------------
__global__ __launch_bounds__(256) void nrm_kernel(const float* __restrict__ src,
                                                  _Float16* __restrict__ dst) {
  const int row = blockIdx.x;
  const int t   = threadIdx.x;
  const float* p = src + (size_t)row * DIM;

  v4f x = *(const v4f*)(p + t * 4);
  float ss = x.x * x.x + x.y * x.y + x.z * x.z + x.w * x.w;

  __shared__ float red[256];
  red[t] = ss;
  __syncthreads();
  for (int s = 128; s > 0; s >>= 1) {
    if (t < s) red[t] += red[t + s];
    __syncthreads();
  }
  const float scale = 1.0f / fmaxf(sqrtf(red[0]), 1e-12f);

  _Float16* q = dst + (size_t)row * DIM + t * 4;
  q[0] = (_Float16)(x.x * scale);
  q[1] = (_Float16)(x.y * scale);
  q[2] = (_Float16)(x.z * scale);
  q[3] = (_Float16)(x.w * scale);
}

// ---------------------------------------------------------------------------
// Kernel 2: fused GEMM (sim = C16 * G16^T * 10) + InfoNCE epilogue.
// Grid: (128 row-strips of 64, COL_SPLIT). Block: 256 threads = 8 waves.
// Wave w: 2M x 2N register block -> mpair = w>>2 (rows mpair*32..+31),
// npair = w&3 (two 16-col tiles at nb + npair*32). 4 accumulators,
// 2 A-frags + 2 B-frags -> 4 WMMAs per K-step (32 B loaded per WMMA).
// Per-row partials land in unique slot [row*8 + split*4 + npair]:
// deterministic, no atomics, every slot written -> no init pass.
// ---------------------------------------------------------------------------
__global__ __launch_bounds__(256) void simloss_kernel(
    const _Float16* __restrict__ C16, const _Float16* __restrict__ G16,
    const int* __restrict__ labels,
    float* __restrict__ sumall_p, float* __restrict__ sumpos_p,
    float* __restrict__ maxn_p) {
  const int strip = blockIdx.x;   // rows [strip*64, strip*64+64)
  const int split = blockIdx.y;   // column half
  const int tid   = threadIdx.x;
  const int wave  = tid >> 5;
  const int lane  = tid & 31;
  const int sub   = lane & 15;    // position within 16-lane group
  const int hi    = lane >> 4;    // lane group (selects K-halves / M-halves)

  const int mpair = wave >> 2;    // 0..1 : rows [mpair*32, +32) within strip
  const int npair = wave & 3;     // 0..3 : two 16-col tiles at nb + npair*32
  const int am0   = strip * M_STRIP + mpair * 32;

  __shared__ int rowlab[M_STRIP];
  if (tid < M_STRIP) rowlab[tid] = labels[strip * M_STRIP + tid];
  __syncthreads();

  // Per-lane persistent accumulators for 16 rows:
  //   idx = mt*8 + r  ->  row = am0 + mt*16 + hi*8 + r
  // (f32 16x16 C/D layout: VGPR r / lanes0-15 -> M=r, lanes16-31 -> M=r+8)
  float sumall[16], sumpos[16], maxn[16];
#pragma unroll
  for (int i = 0; i < 16; ++i) {
    sumall[i] = 0.0f;
    sumpos[i] = 0.0f;
    maxn[i]   = -__builtin_inff();
  }

  // A fragment sources: A 16x32 in contiguous k order (same as B). Lane covers row (am+sub);
  // lanes 0-15 hold K = k..k+15, lanes 16-31 hold K = k+16..k+31.
  const _Float16* Arow0 = C16 + (size_t)(am0 + sub) * DIM;
  const _Float16* Arow1 = C16 + (size_t)(am0 + 16 + sub) * DIM;

  const int colsPerWG = N_ROWS / COL_SPLIT;
  const int col0      = split * colsPerWG;

  for (int nb = col0; nb < col0 + colsPerWG; nb += N_STEP) {
    const int bn0 = nb + npair * 32;  // this wave's two N-tiles: bn0, bn0+16
    // B = G^T : B 32x16 in contiguous k order as A -> lane holds column N=(bn+sub),
    // K-halves per lane group; a contiguous 16-half chunk of G's row.
    const _Float16* Brow0 = G16 + (size_t)(bn0 + sub) * DIM + hi * 16;
    const _Float16* Brow1 = G16 + (size_t)(bn0 + 16 + sub) * DIM + hi * 16;

    v8f acc00 = {}, acc01 = {}, acc10 = {}, acc11 = {};
#pragma unroll 2
    for (int k = 0; k < DIM; k += 32) {
      v16h a0  = *(const v16h*)(Arow0 + k + hi * 16);
      v16h a1  = *(const v16h*)(Arow1 + k + hi * 16);
      v16h b0 = *(const v16h*)(Brow0 + k);
      v16h b1 = *(const v16h*)(Brow1 + k);
      acc00 = __builtin_amdgcn_wmma_f32_16x16x32_f16(false, a0, false, b0,
                                                     (short)0, acc00, false, false);
      acc01 = __builtin_amdgcn_wmma_f32_16x16x32_f16(false, a0, false, b1,
                                                     (short)0, acc01, false, false);
      acc10 = __builtin_amdgcn_wmma_f32_16x16x32_f16(false, a1, false, b0,
                                                     (short)0, acc10, false, false);
      acc11 = __builtin_amdgcn_wmma_f32_16x16x32_f16(false, a1, false, b1,
                                                     (short)0, acc11, false, false);
    }

    // Branchless epilogue: fold this 32x32 sim block into row accumulators.
    const int colA = bn0 + sub;
    const int colB = bn0 + 16 + sub;
    const int labA = labels[colA];
    const int labB = labels[colB];
    const float ninf = -__builtin_inff();
#pragma unroll
    for (int mt = 0; mt < 2; ++mt) {
      const int rowbase = am0 + mt * 16 + hi * 8;
      const int labbase = mpair * 32 + mt * 16 + hi * 8;
#pragma unroll
      for (int r = 0; r < 8; ++r) {
        const int row    = rowbase + r;
        const int rlab   = rowlab[labbase + r];
        const int i      = mt * 8 + r;
        const v8f accA   = mt ? acc10 : acc00;
        const v8f accB   = mt ? acc11 : acc01;
        {
          float sim = accA[r] * TEMP_INV;
          float e   = __expf(sim);
          bool self = (row == colA);
          float em  = self ? 0.0f : e;              // v_cndmask
          sumall[i] += em;
          maxn[i] = fmaxf(maxn[i], self ? ninf : sim);
          sumpos[i] += (rlab == labA) ? em : 0.0f;  // em==0 covers self
        }
        {
          float sim = accB[r] * TEMP_INV;
          float e   = __expf(sim);
          bool self = (row == colB);
          float em  = self ? 0.0f : e;
          sumall[i] += em;
          maxn[i] = fmaxf(maxn[i], self ? ninf : sim);
          sumpos[i] += (rlab == labB) ? em : 0.0f;
        }
      }
    }
  }

  // Reduce the 16 columns held across each 16-lane group (fixed order).
  const int slot = split * 4 + npair;  // unique per row among NSLOTS writers
#pragma unroll
  for (int mt = 0; mt < 2; ++mt) {
    const int rowbase = am0 + mt * 16 + hi * 8;
#pragma unroll
    for (int r = 0; r < 8; ++r) {
      const int i = mt * 8 + r;
      float sa = sumall[i], sp = sumpos[i], mx = maxn[i];
      for (int off = 8; off > 0; off >>= 1) {
        sa += __shfl_xor(sa, off, 16);
        sp += __shfl_xor(sp, off, 16);
        mx = fmaxf(mx, __shfl_xor(mx, off, 16));
      }
      if (sub == 0) {
        const int row = rowbase + r;
        sumall_p[row * NSLOTS + slot] = sa;
        sumpos_p[row * NSLOTS + slot] = sp;
        maxn_p[row * NSLOTS + slot]   = mx;
      }
    }
  }
}

// ---------------------------------------------------------------------------
// Kernel 3: combine NSLOTS partials per row, apply eps/clip/log + fallback.
// ---------------------------------------------------------------------------
__global__ __launch_bounds__(256) void finalize_kernel(
    const float* __restrict__ sa_p, const float* __restrict__ sp_p,
    const float* __restrict__ mx_p, float* __restrict__ rowloss) {
  const int i = blockIdx.x * 256 + threadIdx.x;
  if (i >= N_ROWS) return;
  float sa = 0.0f, sp = 0.0f, mx = -__builtin_inff();
#pragma unroll
  for (int s = 0; s < NSLOTS; ++s) {
    sa += sa_p[i * NSLOTS + s];
    sp += sp_p[i * NSLOTS + s];
    mx = fmaxf(mx, mx_p[i * NSLOTS + s]);
  }
  float frac = (sp + EPSF) / (sa + EPSF);
  frac = fminf(fmaxf(frac, EPSF), 1.0f);
  float loss = -logf(frac);
  // sp accumulates exp(sim) >= e^-10 terms only; sp == 0 exactly <=> no
  // positive exists for this anchor -> fallback to -max off-diagonal sim.
  if (sp == 0.0f) loss = -mx;
  rowloss[i] = loss;
}

// ---------------------------------------------------------------------------
// Kernel 4: deterministic single-block mean of the 8192 row losses.
// ---------------------------------------------------------------------------
__global__ __launch_bounds__(256) void mean_kernel(
    const float* __restrict__ rowloss, float* __restrict__ out) {
  __shared__ float red[256];
  float s = 0.0f;
  for (int i = threadIdx.x; i < N_ROWS; i += 256) s += rowloss[i];
  red[threadIdx.x] = s;
  __syncthreads();
  for (int st = 128; st > 0; st >>= 1) {
    if (threadIdx.x < st) red[threadIdx.x] += red[threadIdx.x + st];
    __syncthreads();
  }
  if (threadIdx.x == 0) out[0] = red[0] / (float)N_ROWS;
}

// ---------------------------------------------------------------------------
extern "C" void kernel_launch(void* const* d_in, const int* in_sizes, int n_in,
                              void* d_out, int out_size, void* d_ws,
                              size_t ws_size, hipStream_t stream) {
  (void)in_sizes; (void)n_in; (void)out_size; (void)ws_size;
  const float* ctx  = (const float*)d_in[0];
  const float* gls  = (const float*)d_in[1];
  const int*   labs = (const int*)d_in[2];

  char* ws = (char*)d_ws;
  _Float16* C16 = (_Float16*)ws;                                   // 16 MB
  _Float16* G16 = (_Float16*)(ws + (size_t)N_ROWS * DIM * 2);      // 16 MB
  float* sa = (float*)(ws + (size_t)N_ROWS * DIM * 4);             // 256 KB
  float* sp = sa + N_ROWS * NSLOTS;                                // 256 KB
  float* mx = sp + N_ROWS * NSLOTS;                                // 256 KB
  float* rl = mx + N_ROWS * NSLOTS;                                // 32 KB

  nrm_kernel<<<N_ROWS, 256, 0, stream>>>(ctx, C16);
  nrm_kernel<<<N_ROWS, 256, 0, stream>>>(gls, G16);

  dim3 grid(N_ROWS / M_STRIP, COL_SPLIT);
  simloss_kernel<<<grid, 256, 0, stream>>>(C16, G16, labs, sa, sp, mx);

  finalize_kernel<<<N_ROWS / 256, 256, 0, stream>>>(sa, sp, mx, rl);
  mean_kernel<<<1, 256, 0, stream>>>(rl, (float*)d_out);
}
